// Eyettention_readerID_7490422964773
// MI455X (gfx1250) — matrix operand order repaired and checked by reading, not compiled, not run on hardware
//
#include <hip/hip_runtime.h>
#include <math.h>

// ---------------- model dims ----------------
#define BATCH 128
#define DBERT 768
#define NSN   60
#define NSP   120
#define TDEC  119      // MAX_SP - 1
#define HDEC  128      // decoder hidden
#define HD    64       // encoder per-direction hidden
#define SUBE  32
#define LSUB  96
#define KDEC  832      // 802 padded to /32
#define NOUT  117      // MAX_SN*2-3

typedef __attribute__((ext_vector_type(16))) _Float16 v16h;
typedef __attribute__((ext_vector_type(8)))  float    v8f;
typedef __attribute__((ext_vector_type(4)))  unsigned int u32x4;
typedef __attribute__((ext_vector_type(8)))  unsigned int u32x8;

// ---------------- Tensor Data Mover: async global->LDS via D# descriptor ----------------
// Builds a minimal 2D D# (one row of nbytes/4 dword elements) per CDNA5 ISA
// group0: count=1 | lds_addr | global_addr[56:0] | type=2 ; group1: data_size=4B, dims/strides.
// Issued as inline asm (portable across the 5-arg/6-arg builtin split); tracked by TENSORcnt.
__device__ __forceinline__ void tdm_load_to_lds(const void* gsrc, void* lds_dst,
                                                unsigned nbytes) {
  unsigned long long ga = (unsigned long long)(uintptr_t)gsrc;
  unsigned lds_off = (unsigned)(uintptr_t)lds_dst;   // addr[31:0] = LDS offset
  unsigned ne = nbytes >> 2;                         // 4-byte elements (fits 16-bit tile_dim0)
  u32x4 g0;
  g0[0] = 1u;                                        // count=1, user descriptor
  g0[1] = lds_off;                                   // lds_addr (bytes)
  g0[2] = (unsigned)ga;                              // global_addr[31:0]
  g0[3] = (unsigned)((ga >> 32) & 0x01ffffffu) | (2u << 30);  // addr[56:32] | type=2
  u32x8 g1;
  g1[0] = 2u << 16;                                  // data_size = 2 -> 4 bytes
  g1[1] = (ne & 0xffffu) << 16;                      // tensor_dim0[15:0]  (bits 63:48)
  g1[2] = ((ne >> 16) & 0xffffu) | (1u << 16);       // tensor_dim0[31:16] | tensor_dim1 = 1
  g1[3] = (ne & 0xffffu) << 16;                      // tile_dim0          (bits 127:112)
  g1[4] = 1u;                                        // tile_dim1 = 1
  g1[5] = ne;                                        // tensor_dim0_stride[31:0]
  g1[6] = (ne >> 16) & 0xffffu;                      // tensor_dim0_stride[47:32]
  g1[7] = 0u;
  u32x4 gz = {0u, 0u, 0u, 0u};                       // groups 2/3: unused dims (zeros)
  asm volatile("tensor_load_to_lds %0, %1, %2, %3"
               :: "s"(g0), "s"(g1), "s"(gz), "s"(gz)
               : "memory");
}

// ---------------- WMMA helpers (CDNA5 wave32, 16x16x32 f16 -> f32) ----------------
// A-fragment per ISA 7.12.2: lanes 0-15 -> M=lane, halves 0..7 = K0..7, 8..15 = K16..23
//                            lanes 16-31 -> M=lane-16, halves 0..7 = K8..15, 8..15 = K24..31
__device__ __forceinline__ v16h load_a_frag(const _Float16* A, int lda) {
  const int lane = threadIdx.x & 31;
  const _Float16* p = A + (lane & 15) * lda;
  const int kb = (lane < 16) ? 0 : 8;
  v16h a;
#pragma unroll
  for (int j = 0; j < 8; ++j) a[j] = p[kb + j];
#pragma unroll
  for (int j = 0; j < 8; ++j) a[8 + j] = p[kb + 16 + j];
  return a;
}

// B-fragment: B is K(32) x N(16); weight stored row-major W[N][K] (we compute x@W.T).
// lanes 0-15 -> N=lane, K=0..7,16..23 ; lanes 16-31 -> N=lane-16, K=8..15,24..31
__device__ __forceinline__ v16h load_b_frag(const _Float16* W, int ldw) {
  const int lane = threadIdx.x & 31;
  const _Float16* p = W + (lane & 15) * ldw + ((lane < 16) ? 0 : 8);
  v16h b;
#pragma unroll
  for (int j = 0; j < 16; ++j) b[j] = p[(j < 8) ? j : j + 8];
  return b;
}

__device__ __forceinline__ v8f wmma32(v16h a, v16h b, v8f c) {
  return __builtin_amdgcn_wmma_f32_16x16x32_f16(false, a, false, b, (short)0, c,
                                                false, false);
}

__device__ __forceinline__ float sigf(float x) { return 1.0f / (1.0f + __expf(-x)); }

// Block-scope GEMM: C[M,N](f32) = A[M,K](f16,ld=K) * W[N,K]^T (+bias0+bias1) (+Cin) ;
// optional ReLU ; optional f16 mirror into Z (ld=ldz). M,N multiples of 16, K of 32.
__device__ void block_gemm(const _Float16* A, int lda,
                           const _Float16* W, int ldw,
                           int M, int N, int K,
                           const float* bias0, const float* bias1,
                           const float* Cin, float* C,
                           int relu, _Float16* Z, int ldz) {
  const int wave   = threadIdx.x >> 5;
  const int nwaves = blockDim.x >> 5;
  const int lane   = threadIdx.x & 31;
  const int nt_n   = N >> 4;
  const int ntiles = (M >> 4) * nt_n;
  for (int tile = wave; tile < ntiles; tile += nwaves) {
    const int mt = tile / nt_n, nt = tile % nt_n;
    v8f acc = {};
    for (int k = 0; k < K; k += 32) {
      v16h a = load_a_frag(A + (mt * 16) * lda + k, lda);
      v16h b = load_b_frag(W + (nt * 16) * ldw + k, ldw);
      acc = wmma32(a, b, acc);
    }
    const int n  = nt * 16 + (lane & 15);
    const int mb = mt * 16 + ((lane < 16) ? 0 : 8);
#pragma unroll
    for (int r = 0; r < 8; ++r) {
      float v = acc[r];
      if (bias0) v += bias0[n];
      if (bias1) v += bias1[n];
      if (Cin)   v += Cin[(mb + r) * N + n];
      if (relu)  v = fmaxf(v, 0.0f);
      C[(mb + r) * N + n] = v;
      if (Z) Z[(mb + r) * ldz + n] = (_Float16)v;
    }
  }
}

// ---------------- prep kernels ----------------
__global__ void k_cvt_f16(_Float16* dst, const float* src, int ro, int co, int ri, int ci) {
  int idx = blockIdx.x * blockDim.x + threadIdx.x;
  if (idx >= ro * co) return;
  int r = idx / co, c = idx % co;
  dst[idx] = (_Float16)((r < ri && c < ci) ? src[r * ci + c] : 0.0f);
}

__global__ void k_pad_b(float* dst, const float* src, int n_out, int n_in) {
  int i = blockIdx.x * blockDim.x + threadIdx.x;
  if (i < n_out) dst[i] = (i < n_in) ? src[i] : 0.0f;
}

// segment-sum pool of sn subword embeddings -> merged f16, layout (t, b, d)
__global__ void k_pool_sn(const float* emb, const int* wid, _Float16* merged) {
  int idx = blockIdx.x * blockDim.x + threadIdx.x;
  if (idx >= BATCH * DBERT) return;
  int b = idx / DBERT, d = idx % DBERT;
  for (int t = 0; t < NSN; ++t) merged[((size_t)t * BATCH + b) * DBERT + d] = (_Float16)0.0f;
  int cur = wid[b * LSUB];
  float acc = 0.0f;
  for (int l = 0; l < LSUB; ++l) {
    int w = wid[b * LSUB + l];
    if (w != cur) {  // word_ids sorted -> each segment contiguous
      merged[((size_t)cur * BATCH + b) * DBERT + d] = (_Float16)acc;
      acc = 0.0f; cur = w;
    }
    acc += emb[((size_t)b * LSUB + l) * DBERT + d];
  }
  merged[((size_t)cur * BATCH + b) * DBERT + d] = (_Float16)acc;
}

__global__ void k_mask_wlen(const _Float16* merged, const float* wlen,
                            float* mask, float* enc_out) {
  int idx = blockIdx.x * blockDim.x + threadIdx.x;
  if (idx >= BATCH * NSN) return;
  int b = idx / NSN, t = idx % NSN;
  const _Float16* p = merged + ((size_t)t * BATCH + b) * DBERT;
  float s = 0.0f;
  for (int d = 0; d < DBERT; ++d) s += (float)p[d];
  mask[b * NSN + t] = (s != 0.0f) ? 1.0f : 0.0f;
  enc_out[((size_t)b * NSN + t) * (HDEC + 1) + HDEC] = wlen[b * NSN + t];
}

// decoder token pool: tmp(t,b,d) = pos_emb[sp_pos] + segsum(word_emb[sp_emd])
__global__ void k_pool_sp(const float* wemb, const float* pemb,
                          const int* sp_emd, const int* sp_pos, const int* wid_sp,
                          float* tmp) {
  int idx = blockIdx.x * blockDim.x + threadIdx.x;
  if (idx >= BATCH * DBERT) return;
  int b = idx / DBERT, d = idx % DBERT;
  for (int t = 0; t < TDEC; ++t)
    tmp[((size_t)t * BATCH + b) * DBERT + d] = pemb[(size_t)sp_pos[b * NSP + t] * DBERT + d];
  int cur = wid_sp[b * NSP];
  float acc = 0.0f;
  for (int l = 0; l < TDEC; ++l) {
    int w = wid_sp[b * NSP + l];
    if (w != cur) {
      tmp[((size_t)cur * BATCH + b) * DBERT + d] += acc;
      acc = 0.0f; cur = w;
    }
    acc += wemb[(size_t)sp_emd[b * NSP + l] * DBERT + d];
  }
  tmp[((size_t)cur * BATCH + b) * DBERT + d] += acc;
}

// LayerNorm over 768 + concat [landing, fix_dur, sub_emb] -> dec_in f16 (t,b,832)
__global__ void __launch_bounds__(256) k_ln_concat(
    const float* tmp, const float* gamma, const float* beta,
    const float* land, const float* fixd, const int* sub_id, const float* sub_tab,
    _Float16* dec_in) {
  __shared__ float red[8];
  const int tb = blockIdx.x;              // t*BATCH + b
  const int t = tb / BATCH, b = tb % BATCH;
  const float* x = tmp + (size_t)tb * DBERT;
  float s = 0.0f;
  for (int d = threadIdx.x; d < DBERT; d += 256) s += x[d];
  for (int o = 16; o > 0; o >>= 1) s += __shfl_xor(s, o, 32);
  if ((threadIdx.x & 31) == 0) red[threadIdx.x >> 5] = s;
  __syncthreads();
  if (threadIdx.x == 0) { float a = 0; for (int i = 0; i < 8; ++i) a += red[i]; red[0] = a / DBERT; }
  __syncthreads();
  const float mu = red[0];
  __syncthreads();
  float v = 0.0f;
  for (int d = threadIdx.x; d < DBERT; d += 256) { float e = x[d] - mu; v += e * e; }
  for (int o = 16; o > 0; o >>= 1) v += __shfl_xor(v, o, 32);
  if ((threadIdx.x & 31) == 0) red[threadIdx.x >> 5] = v;
  __syncthreads();
  if (threadIdx.x == 0) { float a = 0; for (int i = 0; i < 8; ++i) a += red[i]; red[0] = rsqrtf(a / DBERT + 1e-12f); }
  __syncthreads();
  const float rstd = red[0];
  _Float16* o_ = dec_in + (size_t)tb * KDEC;
  for (int d = threadIdx.x; d < DBERT; d += 256)
    o_[d] = (_Float16)(((x[d] - mu) * rstd) * gamma[d] + beta[d]);
  for (int d = DBERT + threadIdx.x; d < KDEC; d += 256) {
    float val;
    if (d == DBERT)            val = land[b * NSP + t];
    else if (d == DBERT + 1)   val = fixd[b * NSP + t];
    else if (d < DBERT + 2 + SUBE) val = sub_tab[sub_id[b] * SUBE + (d - DBERT - 2)];
    else val = 0.0f;
    o_[d] = (_Float16)val;
  }
}

// ---------------- batched WMMA GEMM (time-parallel projections) ----------------
// each wave owns a 16(M) x 128(N) strip: A-fragment reused across 8 N-tiles
__global__ void __launch_bounds__(256) k_gemm(const _Float16* A, const _Float16* W,
                                              float* C, int M, int N, int K,
                                              const float* b0, const float* b1) {
  const int wave = blockIdx.x * (blockDim.x >> 5) + (threadIdx.x >> 5);
  const int lane = threadIdx.x & 31;
  const int sn = N >> 7;
  if (wave >= (M >> 4) * sn) return;
  const int mt = wave / sn;
  const int n0 = (wave % sn) * 128;
  v8f acc[8];
  v8f zero = {};
#pragma unroll
  for (int j = 0; j < 8; ++j) acc[j] = zero;
  for (int k = 0; k < K; k += 32) {
    v16h a = load_a_frag(A + (size_t)(mt * 16) * K + k, K);
#pragma unroll
    for (int j = 0; j < 8; ++j) {
      v16h b = load_b_frag(W + (size_t)(n0 + j * 16) * K + k, K);
      acc[j] = wmma32(a, b, acc[j]);
    }
  }
  const int mb = mt * 16 + ((lane < 16) ? 0 : 8);
  const int nn = lane & 15;
#pragma unroll
  for (int j = 0; j < 8; ++j) {
    int n = n0 + j * 16 + nn;
    float badd = (b0 ? b0[n] : 0.0f) + (b1 ? b1[n] : 0.0f);
#pragma unroll
    for (int r = 0; r < 8; ++r) C[(size_t)(mb + r) * N + n] = acc[j][r] + badd;
  }
}

// ---------------- persistent encoder BiLSTM layer (one WG per direction) ----------------
// Whh staged into LDS once via TDM; B-fragments then come from LDS (ds_load path).
__global__ void __launch_bounds__(256) k_enc_rec(const float* xproj_f, const float* xproj_b,
                                                 const _Float16* whh_f, const _Float16* whh_b,
                                                 _Float16* xout, float* enc_out) {
  extern __shared__ char smem[];
  _Float16* sh = (_Float16*)smem;                                    // h: 128x64 f16
  float* sc = (float*)(smem + BATCH * HD * 2);                       // c: 128x64 f32
  float* sg = (float*)(smem + BATCH * HD * 2 + BATCH * HD * 4);      // gates: 128x256 f32
  _Float16* swhh = (_Float16*)(smem + BATCH * HD * 2 + BATCH * HD * 4 +
                               BATCH * 256 * 4);                     // Whh: 256x64 f16
  const int dir = blockIdx.x;             // 0 = forward, 1 = backward
  const float* xp = dir ? xproj_b : xproj_f;
  const _Float16* whh = dir ? whh_b : whh_f;
  const int ooff = dir ? HD : 0;          // concat([fwd, bwd]) along feature dim
  const int wave = threadIdx.x >> 5, lane = threadIdx.x & 31;
  if (wave == 0) {                        // TDM: stage Whh into LDS (32KB), wave 0 issues
    tdm_load_to_lds(whh, swhh, 256 * HD * (unsigned)sizeof(_Float16));
    __builtin_amdgcn_s_wait_tensorcnt(0);
  }
  for (int i = threadIdx.x; i < BATCH * HD; i += blockDim.x) { sh[i] = (_Float16)0.0f; sc[i] = 0.0f; }
  __syncthreads();
  for (int step = 0; step < NSN; ++step) {
    const int t = dir ? (NSN - 1 - step) : step;
    const float* g_in = xp + (size_t)t * BATCH * 256;       // x@Wih + bih + bhh (precomputed)
    for (int i = threadIdx.x; i < BATCH * 256; i += blockDim.x) sg[i] = g_in[i];
    __syncthreads();
    // gates += h @ Whh^T  (M=128, N=256, K=64): 128 tiles over 8 waves, B from LDS
    for (int tile = wave; tile < 8 * 16; tile += 8) {
      int mt = tile >> 4, nt = tile & 15;
      v8f acc = {};
      for (int k = 0; k < HD; k += 32) {
        v16h a = load_a_frag(sh + (mt * 16) * HD + k, HD);
        v16h b = load_b_frag(swhh + (nt * 16) * HD + k, HD);
        acc = wmma32(a, b, acc);
      }
      int n = nt * 16 + (lane & 15);
      int mb = mt * 16 + ((lane < 16) ? 0 : 8);
#pragma unroll
      for (int r = 0; r < 8; ++r) sg[(mb + r) * 256 + n] += acc[r];
    }
    __syncthreads();
    for (int i = threadIdx.x; i < BATCH * HD; i += blockDim.x) {
      int b = i / HD, j = i % HD;
      float gi = sg[b * 256 + j],       gf = sg[b * 256 + HD + j];
      float gg = sg[b * 256 + 2 * HD + j], go = sg[b * 256 + 3 * HD + j];
      float cc = sigf(gf) * sc[i] + sigf(gi) * tanhf(gg);
      float hh = sigf(go) * tanhf(cc);
      sc[i] = cc; sh[i] = (_Float16)hh;
      xout[((size_t)t * BATCH + b) * HDEC + ooff + j] = (_Float16)hh;
      if (enc_out) enc_out[((size_t)b * NSN + t) * (HDEC + 1) + ooff + j] = hh;
    }
    __syncthreads();
  }
}

// ---------------- persistent decoder (one 1024-thread WG) ----------------
// Whh[k] (128KB f16) double-buffered in LDS: TDM prefetches cell k+1's weights while
// WMMA consumes cell k's out of the other buffer (s_wait_tensorcnt(1) retires the older).
struct DecParams {
  const float* xproj0;               // (119,128,512) = dec_in@Wih0 + bih0 + bhh0
  const _Float16* wih[8];            // wih[k>=1]: (512,128) f16
  const _Float16* whh[8];            // (512,128) f16
  const float* bih[8];
  const float* bhh[8];
  const float* attn_w;               // (129,128)
  const float* attn_b;               // (129)
  const float* enc_out;              // (128,60,129)
  const float* sn_mask;              // (128,60)
  const int*   sp_pos;               // (128,120)
  const _Float16* dw[5];             // dense f16, K/N padded
  const float* db[5];
  float* gates;                      // (128,512)
  _Float16* h;                       // (8,128,128) f16
  float* c;                          // (8,128,128)
  float* q;                          // (128,129)
  float* att;                        // (128,60)
  _Float16* z0;                      // (128,288)
  _Float16* z1;                      // (128,512)
  _Float16* z2;                      // (128,256)
  _Float16* z3;                      // (128,256)
  float* cbuf;                       // (128,512)
  float* outs;                       // (128,119,117)
  float* atts;                       // (128,119,60)
};

#define WELEM (512 * HDEC)           // elements per Whh buffer

__global__ void __launch_bounds__(1024) k_decoder(DecParams P) {
  extern __shared__ char smem[];
  _Float16* wbuf = (_Float16*)smem;  // 2 x 512*128 f16 = 256KB
  const int tid = threadIdx.x;
  for (int i = tid; i < 8 * BATCH * HDEC; i += 1024) { P.h[i] = (_Float16)0.0f; P.c[i] = 0.0f; }
  if (tid < 32)                      // prologue: stage Whh[0]
    tdm_load_to_lds(P.whh[0], wbuf, WELEM * (unsigned)sizeof(_Float16));
  __syncthreads();
  const _Float16* h7 = P.h + (size_t)7 * BATCH * HDEC;
  for (int t = 0; t < TDEC; ++t) {
    // ---- 8 stacked LSTM cells ----
    for (int k = 0; k < 8; ++k) {
      _Float16* wcur = wbuf + (size_t)(k & 1) * WELEM;
      _Float16* wnxt = wbuf + (size_t)((k + 1) & 1) * WELEM;
      if (tid < 32) {                // wave 0: prefetch next cell's Whh, retire current
        if (k < 7) {
          tdm_load_to_lds(P.whh[k + 1], wnxt, WELEM * (unsigned)sizeof(_Float16));
          __builtin_amdgcn_s_wait_tensorcnt(1);
        } else if (t < TDEC - 1) {
          tdm_load_to_lds(P.whh[0], wnxt, WELEM * (unsigned)sizeof(_Float16));
          __builtin_amdgcn_s_wait_tensorcnt(1);
        } else {
          __builtin_amdgcn_s_wait_tensorcnt(0);
        }
      }
      __syncthreads();
      __builtin_prefetch((const void*)P.wih[k], 0, 3);   // global_prefetch_b8 (L2-resident Wih)
      if (k == 0) {
        block_gemm(P.h, HDEC, wcur, HDEC, BATCH, 512, HDEC,
                   nullptr, nullptr, P.xproj0 + (size_t)t * BATCH * 512, P.gates,
                   0, nullptr, 0);
      } else {
        block_gemm(P.h + (size_t)(k - 1) * BATCH * HDEC, HDEC, P.wih[k], HDEC,
                   BATCH, 512, HDEC, P.bih[k], P.bhh[k], nullptr, P.gates, 0, nullptr, 0);
        __threadfence_block(); __syncthreads();
        block_gemm(P.h + (size_t)k * BATCH * HDEC, HDEC, wcur, HDEC,
                   BATCH, 512, HDEC, nullptr, nullptr, P.gates, P.gates, 0, nullptr, 0);
      }
      __threadfence_block(); __syncthreads();
      float* ck = P.c + (size_t)k * BATCH * HDEC;
      _Float16* hk = P.h + (size_t)k * BATCH * HDEC;
      for (int i = tid; i < BATCH * HDEC; i += 1024) {
        int b = i / HDEC, j = i % HDEC;
        const float* g = P.gates + b * 512;
        float cc = sigf(g[HDEC + j]) * ck[i] + sigf(g[j]) * tanhf(g[2 * HDEC + j]);
        float hh = sigf(g[3 * HDEC + j]) * tanhf(cc);
        ck[i] = cc; hk[i] = (_Float16)hh;
      }
      __threadfence_block(); __syncthreads();
    }
    // ---- cross attention: q = h7 @ attn_w^T + b ----
    for (int i = tid; i < BATCH * 129; i += 1024) {
      int b = i / 129, n = i % 129;
      const float* w = P.attn_w + n * HDEC;
      const _Float16* hb = h7 + b * HDEC;
      float s = P.attn_b[n];
      for (int d = 0; d < HDEC; ++d) s += (float)hb[d] * w[d];
      P.q[i] = s;
    }
    __threadfence_block(); __syncthreads();
    for (int i = tid; i < BATCH * NSN; i += 1024) {
      int b = i / NSN, l = i % NSN;
      const float* eo = P.enc_out + ((size_t)b * NSN + l) * (HDEC + 1);
      const float* qb = P.q + b * 129;
      float s = 0.0f;
      for (int d = 0; d < 129; ++d) s += qb[d] * eo[d];
      int pos = P.sp_pos[b * NSP + t];
      int left = (pos - 1 > 0) ? pos - 1 : 0;
      int right = (pos + 1 < NSN - 1) ? pos + 1 : NSN - 1;
      bool outside = (l < left) || (l > right);
      if (P.sn_mask[b * NSN + l] == 0.0f || outside) s += -1.0e9f;
      P.att[i] = s;
    }
    __threadfence_block(); __syncthreads();
    // windowed softmax * gaussian, per batch row, wave32 shfl reductions
    {
      int wave = tid >> 5, lane = tid & 31;
      for (int bb = wave; bb < BATCH; bb += 32) {
        float v0 = (lane < NSN) ? P.att[bb * NSN + lane] : -3.0e38f;
        float v1 = (lane + 32 < NSN) ? P.att[bb * NSN + lane + 32] : -3.0e38f;
        float mx = fmaxf(v0, v1);
        for (int o = 16; o > 0; o >>= 1) mx = fmaxf(mx, __shfl_xor(mx, o, 32));
        float e0 = (lane < NSN) ? __expf(v0 - mx) : 0.0f;
        float e1 = (lane + 32 < NSN) ? __expf(v1 - mx) : 0.0f;
        float sm = e0 + e1;
        for (int o = 16; o > 0; o >>= 1) sm += __shfl_xor(sm, o, 32);
        float inv = 1.0f / sm;
        int pos = P.sp_pos[bb * NSP + t];
        if (lane < NSN) {
          float dlt = (float)(lane - pos);
          float a = e0 * inv * __expf(-2.0f * dlt * dlt);   // WIN=1 -> exp(-d^2/0.5)
          P.att[bb * NSN + lane] = a;
          P.atts[((size_t)bb * TDEC + t) * NSN + lane] = a;
        }
        if (lane + 32 < NSN) {
          float dlt = (float)(lane + 32 - pos);
          float a = e1 * inv * __expf(-2.0f * dlt * dlt);
          P.att[bb * NSN + lane + 32] = a;
          P.atts[((size_t)bb * TDEC + t) * NSN + lane + 32] = a;
        }
      }
    }
    __threadfence_block(); __syncthreads();
    // ctx = att @ enc_out ; z0 = concat(ctx, h7) padded to 288 (f16)
    for (int i = tid; i < BATCH * 288; i += 1024) {
      int b = i / 288, d = i % 288;
      float v;
      if (d < 129) {
        const float* ab = P.att + b * NSN;
        const float* eo = P.enc_out + (size_t)b * NSN * (HDEC + 1) + d;
        float s = 0.0f;
        for (int l = 0; l < NSN; ++l) s += ab[l] * eo[l * (HDEC + 1)];
        v = s;
      } else if (d < 257) {
        v = (float)h7[b * HDEC + (d - 129)];
      } else v = 0.0f;
      P.z0[i] = (_Float16)v;
    }
    __threadfence_block(); __syncthreads();
    // ---- dense head (WMMA) ----
    block_gemm(P.z0, 288, P.dw[0], 288, BATCH, 512, 288, P.db[0], nullptr, nullptr, P.cbuf, 1, P.z1, 512);
    __threadfence_block(); __syncthreads();
    block_gemm(P.z1, 512, P.dw[1], 512, BATCH, 256, 512, P.db[1], nullptr, nullptr, P.cbuf, 1, P.z2, 256);
    __threadfence_block(); __syncthreads();
    block_gemm(P.z2, 256, P.dw[2], 256, BATCH, 256, 256, P.db[2], nullptr, nullptr, P.cbuf, 1, P.z3, 256);
    __threadfence_block(); __syncthreads();
    block_gemm(P.z3, 256, P.dw[3], 256, BATCH, 256, 256, P.db[3], nullptr, nullptr, P.cbuf, 1, P.z2, 256);
    __threadfence_block(); __syncthreads();
    block_gemm(P.z2, 256, P.dw[4], 256, BATCH, 128, 256, P.db[4], nullptr, nullptr, P.cbuf, 0, nullptr, 0);
    __threadfence_block(); __syncthreads();
    for (int i = tid; i < BATCH * NOUT; i += 1024) {
      int b = i / NOUT, n = i % NOUT;
      P.outs[((size_t)b * TDEC + t) * NOUT + n] = P.cbuf[b * 128 + n];
    }
    __builtin_amdgcn_s_cluster_barrier();   // NOP unclustered; hook for WGP-cluster scaling
    __syncthreads();
  }
}

// ---------------- host ----------------
extern "C" void kernel_launch(void* const* d_in, const int* in_sizes, int n_in,
                              void* d_out, int out_size, void* d_ws, size_t ws_size,
                              hipStream_t stream) {
  (void)in_sizes; (void)n_in; (void)out_size; (void)ws_size;
  // Input order: pytree flatten with dict keys sorted ('bwd' before 'fwd').
  // 0..13 leading tensors; 14..77 enc_lstm (8 layers x [bwd(Wih,Whh,bih,bhh), fwd(...)]);
  // 78..109 dec_cells (8 x 4); 110 attn_w; 111 attn_b; 112..121 dense (5 x (W,b)).
  const float* sn_emb   = (const float*)d_in[0];
  const int*   wid_sn   = (const int*)d_in[1];
  const float* sn_wlen  = (const float*)d_in[2];
  const int*   sp_emd   = (const int*)d_in[3];
  const int*   sp_pos   = (const int*)d_in[4];
  const int*   wid_sp   = (const int*)d_in[5];
  const float* sp_fix   = (const float*)d_in[6];
  const float* sp_land  = (const float*)d_in[7];
  const int*   sub_id   = (const int*)d_in[8];
  const float* word_emb = (const float*)d_in[9];
  const float* pos_emb  = (const float*)d_in[10];
  const float* ln_g     = (const float*)d_in[11];
  const float* ln_b     = (const float*)d_in[12];
  const float* sub_tab  = (const float*)d_in[13];

  char* wsb = (char*)d_ws;
  size_t off = 0;
  auto alloc = [&](size_t bytes) -> void* {
    off = (off + 255) & ~(size_t)255;
    void* p = wsb + off;
    off += bytes;
    return p;
  };

  _Float16* merged16 = (_Float16*)alloc((size_t)NSN * BATCH * DBERT * 2);
  float* sn_mask = (float*)alloc((size_t)BATCH * NSN * 4);
  float* enc_out = (float*)alloc((size_t)BATCH * NSN * (HDEC + 1) * 4);
  _Float16* xbuf0 = (_Float16*)alloc((size_t)NSN * BATCH * HDEC * 2);
  _Float16* xbuf1 = (_Float16*)alloc((size_t)NSN * BATCH * HDEC * 2);
  float* xprojF = (float*)alloc((size_t)NSN * BATCH * 256 * 4);
  float* xprojB = (float*)alloc((size_t)NSN * BATCH * 256 * 4);
  float* dec_tmp = (float*)alloc((size_t)TDEC * BATCH * DBERT * 4);
  _Float16* dec_in = (_Float16*)alloc((size_t)TDEC * BATCH * KDEC * 2);
  float* xproj0 = (float*)alloc((size_t)TDEC * BATCH * 512 * 4);
  float* gates = (float*)alloc((size_t)BATCH * 512 * 4);
  _Float16* h_ws = (_Float16*)alloc((size_t)8 * BATCH * HDEC * 2);
  float* c_ws = (float*)alloc((size_t)8 * BATCH * HDEC * 4);
  float* q_ws = (float*)alloc((size_t)BATCH * 129 * 4);
  float* att_ws = (float*)alloc((size_t)BATCH * NSN * 4);
  _Float16* z0 = (_Float16*)alloc((size_t)BATCH * 288 * 2);
  _Float16* z1 = (_Float16*)alloc((size_t)BATCH * 512 * 2);
  _Float16* z2 = (_Float16*)alloc((size_t)BATCH * 256 * 2);
  _Float16* z3 = (_Float16*)alloc((size_t)BATCH * 256 * 2);
  float* cbuf = (float*)alloc((size_t)BATCH * 512 * 4);
  float* b4pad = (float*)alloc(128 * 4);

  auto cvt = [&](const void* src, _Float16* dst, int ro, int co, int ri, int ci) {
    int tot = ro * co;
    k_cvt_f16<<<(tot + 255) / 256, 256, 0, stream>>>(dst, (const float*)src, ro, co, ri, ci);
  };

  // fp16 weight pool
  _Float16* encWih16[8][2];  // [layer][0=fwd,1=bwd]
  _Float16* encWhh16[8][2];
  const float *encBihF[8], *encBhhF[8], *encBihB[8], *encBhhB[8];
  for (int l = 0; l < 8; ++l) {
    int K = (l == 0) ? DBERT : HDEC;
    int eb = 14 + l * 8;       // bwd tuple base
    int ef = eb + 4;           // fwd tuple base
    encWih16[l][0] = (_Float16*)alloc((size_t)256 * K * 2);
    encWih16[l][1] = (_Float16*)alloc((size_t)256 * K * 2);
    encWhh16[l][0] = (_Float16*)alloc((size_t)256 * HD * 2);
    encWhh16[l][1] = (_Float16*)alloc((size_t)256 * HD * 2);
    cvt(d_in[ef + 0], encWih16[l][0], 256, K, 256, K);
    cvt(d_in[eb + 0], encWih16[l][1], 256, K, 256, K);
    cvt(d_in[ef + 1], encWhh16[l][0], 256, HD, 256, HD);
    cvt(d_in[eb + 1], encWhh16[l][1], 256, HD, 256, HD);
    encBihF[l] = (const float*)d_in[ef + 2]; encBhhF[l] = (const float*)d_in[ef + 3];
    encBihB[l] = (const float*)d_in[eb + 2]; encBhhB[l] = (const float*)d_in[eb + 3];
  }
  _Float16* decWih16[8];
  _Float16* decWhh16[8];
  decWih16[0] = (_Float16*)alloc((size_t)512 * KDEC * 2);
  cvt(d_in[78 + 0], decWih16[0], 512, KDEC, 512, 802);
  for (int k = 1; k < 8; ++k) {
    decWih16[k] = (_Float16*)alloc((size_t)512 * HDEC * 2);
    cvt(d_in[78 + k * 4 + 0], decWih16[k], 512, HDEC, 512, HDEC);
  }
  for (int k = 0; k < 8; ++k) {
    decWhh16[k] = (_Float16*)alloc((size_t)512 * HDEC * 2);
    cvt(d_in[78 + k * 4 + 1], decWhh16[k], 512, HDEC, 512, HDEC);
  }
  _Float16* dw16[5];
  const int dKp[5] = {288, 512, 256, 256, 256};
  const int dKi[5] = {257, 512, 256, 256, 256};
  const int dNp[5] = {512, 256, 256, 256, 128};
  const int dNi[5] = {512, 256, 256, 256, 117};
  for (int i = 0; i < 5; ++i) {
    dw16[i] = (_Float16*)alloc((size_t)dNp[i] * dKp[i] * 2);
    cvt(d_in[112 + 2 * i], dw16[i], dNp[i], dKp[i], dNi[i], dKi[i]);
  }
  k_pad_b<<<1, 128, 0, stream>>>(b4pad, (const float*)d_in[113 + 2 * 4], 128, 117);

  // ---- prep ----
  k_pool_sn<<<(BATCH * DBERT + 255) / 256, 256, 0, stream>>>(sn_emb, wid_sn, merged16);
  k_mask_wlen<<<(BATCH * NSN + 255) / 256, 256, 0, stream>>>(merged16, sn_wlen, sn_mask, enc_out);
  k_pool_sp<<<(BATCH * DBERT + 255) / 256, 256, 0, stream>>>(word_emb, pos_emb, sp_emd, sp_pos,
                                                             wid_sp, dec_tmp);
  k_ln_concat<<<TDEC * BATCH, 256, 0, stream>>>(dec_tmp, ln_g, ln_b, sp_land, sp_fix,
                                                sub_id, sub_tab, dec_in);

  // ---- encoder: per layer, batched x@Wih GEMM then persistent recurrence ----
  const _Float16* xin = merged16;
  int K = DBERT;
  const size_t enc_smem = (size_t)BATCH * HD * 2 + (size_t)BATCH * HD * 4 +
                          (size_t)BATCH * 256 * 4 + (size_t)256 * HD * 2;
  for (int l = 0; l < 8; ++l) {
    int nstrips = (NSN * BATCH / 16) * (256 / 128);
    k_gemm<<<(nstrips + 7) / 8, 256, 0, stream>>>(xin, encWih16[l][0], xprojF,
                                                  NSN * BATCH, 256, K, encBihF[l], encBhhF[l]);
    k_gemm<<<(nstrips + 7) / 8, 256, 0, stream>>>(xin, encWih16[l][1], xprojB,
                                                  NSN * BATCH, 256, K, encBihB[l], encBhhB[l]);
    _Float16* xout = (l & 1) ? xbuf1 : xbuf0;
    k_enc_rec<<<2, 256, enc_smem, stream>>>(xprojF, xprojB, encWhh16[l][0], encWhh16[l][1],
                                            xout, (l == 7) ? enc_out : nullptr);
    xin = xout;
    K = HDEC;
  }

  // ---- decoder cell-0 input projection (time-parallel) ----
  {
    int nstrips = (TDEC * BATCH / 16) * (512 / 128);
    k_gemm<<<(nstrips + 7) / 8, 256, 0, stream>>>(dec_in, decWih16[0], xproj0,
                                                  TDEC * BATCH, 512, KDEC,
                                                  (const float*)d_in[78 + 2],
                                                  (const float*)d_in[78 + 3]);
  }

  // ---- persistent decoder ----
  DecParams P;
  P.xproj0 = xproj0;
  for (int k = 0; k < 8; ++k) {
    P.wih[k] = decWih16[k];
    P.whh[k] = decWhh16[k];
    P.bih[k] = (const float*)d_in[78 + k * 4 + 2];
    P.bhh[k] = (const float*)d_in[78 + k * 4 + 3];
  }
  P.attn_w = (const float*)d_in[110];
  P.attn_b = (const float*)d_in[111];
  P.enc_out = enc_out;
  P.sn_mask = sn_mask;
  P.sp_pos = sp_pos;
  for (int i = 0; i < 5; ++i) P.dw[i] = dw16[i];
  for (int i = 0; i < 4; ++i) P.db[i] = (const float*)d_in[113 + 2 * i];
  P.db[4] = b4pad;
  P.gates = gates; P.h = h_ws; P.c = c_ws; P.q = q_ws; P.att = att_ws;
  P.z0 = z0; P.z1 = z1; P.z2 = z2; P.z3 = z3; P.cbuf = cbuf;
  float* out_f = (float*)d_out;
  P.outs = out_f;
  P.atts = out_f + (size_t)BATCH * TDEC * NOUT;
  k_decoder<<<1, 1024, 2 * WELEM * sizeof(_Float16), stream>>>(P);
}
